// AttentionFlow_9758165696821
// MI455X (gfx1250) — matrix operand order repaired and checked by reading, not compiled, not run on hardware
//
#include <hip/hip_runtime.h>
#include <hip/hip_bf16.h>

typedef __attribute__((ext_vector_type(16))) _Float16 v16h;
typedef __attribute__((ext_vector_type(8)))  _Float16 v8h;
typedef __attribute__((ext_vector_type(8)))  float    v8f;

#define TD 8192
#define JD 1024
#define DD 256
#define QS 288   // padded K (f16 query row stride): 256 data + [256]=qbias,[257]=1, rest 0
#define CMS 296  // ctxm LDS row stride (halves): >=288, 16B-aligned rows, bank-conflict-free

// A-fragment (16x32 f16): per lane, elems 0-7 = K off..off+7, elems 8-15 = K off+16..off+23, off=(lane>=16)?8:0.
// p is pre-offset by half*8; load [p,p+8) and [p+16,p+24).
__device__ __forceinline__ v16h load_afrag(const _Float16* p) {
    union { v16h v; v8h h[2]; } u;
    u.h[0] = *(const v8h*)(p);
    u.h[1] = *(const v8h*)(p + 16);
    return u.v;
}
// B-fragment (32x16 f16): per lane (n = lane&15), elems 0-7 = K off..off+7, elems 8-15 = K off+16..off+23, off=half*8.
// p is pre-offset by half*8.
__device__ __forceinline__ v16h load_bfrag(const _Float16* p) {
    union { v16h v; v8h h[2]; } u;
    u.h[0] = *(const v8h*)(p);
    u.h[1] = *(const v8h*)(p + 16);
    return u.v;
}

// ---------------- prep: query -> f16 (row-major, K-padded with bias cols) + transposed ------------
__global__ void af_prep(const float* __restrict__ query, const float* __restrict__ w,
                        _Float16* __restrict__ qf16, _Float16* __restrict__ qTf16)
{
    __shared__ float red[DD];
    const int j = blockIdx.x;
    const int d = threadIdx.x;
    float x = query[(size_t)j * DD + d];
    qf16[(size_t)j * QS + d]  = (_Float16)x;
    qTf16[(size_t)d * JD + j] = (_Float16)x;
    red[d] = x * w[DD + d];                          // qbias partial
    __syncthreads();
    for (int s = DD / 2; s > 0; s >>= 1) {
        if (d < s) red[d] += red[d + s];
        __syncthreads();
    }
    if (d < QS - DD) {                               // fill K-pad columns 256..287
        _Float16 v = (_Float16)0.f;
        if (d == 0) v = (_Float16)red[0];            // col 256: qbias[j]  (pairs with ctxm col 256 = 1)
        if (d == 1) v = (_Float16)1.f;               // col 257: 1         (pairs with ctxm col 257 = cbias[m])
        qf16[(size_t)j * QS + DD + d] = v;
    }
}

// ---------------- phase 1: fused S^T GEMM + 64j-blocked online softmax + P@query GEMM -------------
__launch_bounds__(32)
__global__ void af_phase1(const float* __restrict__ ctx,
                          const float* __restrict__ w,
                          const _Float16* __restrict__ qf16,
                          const _Float16* __restrict__ qTf16,
                          float* __restrict__ G,
                          float* __restrict__ b_buf)
{
    __shared__ __align__(16) float    ctx_s[16][260];
    __shared__ __align__(16) _Float16 ctxm_s[16][CMS];
    __shared__ float cbias_s[16];

    const int lane = threadIdx.x;
    const int half = lane >> 4;
    const int l16  = lane & 15;
    const int row0 = blockIdx.x * 16;

    if (lane < 16) cbias_s[lane] = 0.f;
    __syncthreads();

    // Load 16x256 context tile; build f16 (context*w_m); accumulate cbias = ctx . w_c
    for (int i = lane; i < 16 * 64; i += 32) {
        const int r = i >> 6, c = (i & 63) * 4;
        const float4 cv = *(const float4*)(ctx + (size_t)(row0 + r) * DD + c);
        const float4 wc = *(const float4*)(w + c);
        const float4 wm = *(const float4*)(w + 2 * DD + c);
        *(float4*)&ctx_s[r][c] = cv;
        _Float16* cm = &ctxm_s[r][c];
        cm[0] = (_Float16)(cv.x * wm.x); cm[1] = (_Float16)(cv.y * wm.y);
        cm[2] = (_Float16)(cv.z * wm.z); cm[3] = (_Float16)(cv.w * wm.w);
        atomicAdd(&cbias_s[r], cv.x * wc.x + cv.y * wc.y + cv.z * wc.z + cv.w * wc.w);
    }
    __syncthreads();
    if (lane < 16) {                                  // bias columns of ctxm: [256]=1, [257]=cbias[m]
        for (int c = DD; c < QS; ++c) ctxm_s[lane][c] = (_Float16)0.f;
        ctxm_s[lane][DD]     = (_Float16)1.f;
        ctxm_s[lane][DD + 1] = (_Float16)cbias_s[lane];
    }
    __syncthreads();

    v8f acc[16];                       // c2q accumulator: d-tile t, row m = r + 8*half, col d = t*16 + l16
#pragma unroll
    for (int t = 0; t < 16; ++t)
#pragma unroll
        for (int e = 0; e < 8; ++e) acc[t][e] = 0.f;

    float M = -1e30f, L = 0.f;         // running row max/sum for row m = l16 (replicated in both halves)

    for (int j0 = 0; j0 < JD; j0 += 64) {
        v8f s[4];                      // 2 chunks of 32 j -> 4 S^T tiles (rows j, col m = l16)
#pragma unroll
        for (int q = 0; q < 4; ++q)
#pragma unroll
            for (int e = 0; e < 8; ++e) s[q][e] = 0.f;

        // S^T += query(64x288) @ ctxm^T(288x16)  -- biases ride along in K columns 256/257
#pragma unroll
        for (int kk = 0; kk < 9; ++kk) {
            const v16h bc = load_bfrag(&ctxm_s[l16][kk * 32 + half * 8]);
#pragma unroll
            for (int c = 0; c < 2; ++c) {
                const v16h a0 = load_afrag(qf16 + (size_t)(j0 + c * 32 + l16) * QS + kk * 32 + half * 8);
                s[2 * c] = __builtin_amdgcn_wmma_f32_16x16x32_f16(false, a0, false, bc, (short)0, s[2 * c], false, false);
                const v16h a1 = load_afrag(qf16 + (size_t)(j0 + c * 32 + 16 + l16) * QS + kk * 32 + half * 8);
                s[2 * c + 1] = __builtin_amdgcn_wmma_f32_16x16x32_f16(false, a1, false, bc, (short)0, s[2 * c + 1], false, false);
            }
        }

        // block row-max over 64 j (per lane: same row m for all 32 values)
        float tmax = -1e30f;
#pragma unroll
        for (int q = 0; q < 4; ++q)
#pragma unroll
            for (int r = 0; r < 8; ++r) tmax = fmaxf(tmax, s[q][r]);
        tmax = fmaxf(tmax, __shfl_xor(tmax, 16, 32));
        const float mnew  = fmaxf(M, tmax);
        const float scale = __expf(M - mnew);
        M = mnew;
        L *= scale;

        // rescale accumulator (row of reg r is m = r + 8*half; stats live at lane m)
        float sc[8];
#pragma unroll
        for (int r = 0; r < 8; ++r) sc[r] = __shfl(scale, r + 8 * half, 32);
#pragma unroll
        for (int t = 0; t < 16; ++t)
#pragma unroll
            for (int r = 0; r < 8; ++r) acc[t][r] *= sc[r];

        // exp -> P (A-fragment layout, zero data movement) + PV GEMM, per 32-j chunk
        float psum = 0.f;
#pragma unroll
        for (int c = 0; c < 2; ++c) {
            union { v16h v; _Float16 e[16]; } P;
#pragma unroll
            for (int r = 0; r < 8; ++r) {
                const float p0 = __expf(s[2 * c][r] - M);
                const float p1 = __expf(s[2 * c + 1][r] - M);
                psum += p0 + p1;
                P.e[r]     = (_Float16)p0;
                P.e[r + 8] = (_Float16)p1;
            }
#pragma unroll
            for (int t = 0; t < 16; ++t) {
                const v16h b = load_bfrag(qTf16 + (size_t)(t * 16 + l16) * JD + j0 + c * 32 + half * 8);
                acc[t] = __builtin_amdgcn_wmma_f32_16x16x32_f16(false, P.v, false, b, (short)0, acc[t], false, false);
            }
        }
        psum += __shfl_xor(psum, 16, 32);
        L += psum;
    }

    if (lane < 16) b_buf[row0 + lane] = M;            // b = rowmax(S), biases included

    const float linv = 1.0f / L;
    float li[8];
#pragma unroll
    for (int r = 0; r < 8; ++r) li[r] = __shfl(linv, r + 8 * half, 32);

    // G[:, 256:512] = c2q ; G[:, 512:768] = context * c2q
#pragma unroll
    for (int t = 0; t < 16; ++t) {
        const int d = t * 16 + l16;
#pragma unroll
        for (int r = 0; r < 8; ++r) {
            const int m = r + 8 * half;
            const float c2q = acc[t][r] * li[r];
            const float cx  = ctx_s[m][d];
            float* Gr = G + (size_t)(row0 + m) * (4 * DD);
            Gr[DD + d]     = c2q;
            Gr[2 * DD + d] = cx * c2q;
        }
    }
    // G[:, 0:256] = context
    for (int i = lane; i < 16 * 64; i += 32) {
        const int r = i >> 6, c = (i & 63) * 4;
        const float4 v = *(const float4*)&ctx_s[r][c];
        *(float4*)(G + (size_t)(row0 + r) * (4 * DD) + c) = v;
    }
}

// ---------------- h_tilde = b @ context (two-stage deterministic reduction) -----------------------
__global__ void af_hpart(const float* __restrict__ ctx, const float* __restrict__ b_buf,
                         float* __restrict__ partials)
{
    const int d  = threadIdx.x;
    const int r0 = blockIdx.x * 128;
    float s = 0.f;
    for (int r = 0; r < 128; ++r)
        s += b_buf[r0 + r] * ctx[(size_t)(r0 + r) * DD + d];
    partials[blockIdx.x * DD + d] = s;
}

__global__ void af_hreduce(const float* __restrict__ partials, float* __restrict__ h)
{
    const int d = threadIdx.x;
    float s = 0.f;
    for (int b = 0; b < TD / 128; ++b) s += partials[b * DD + d];
    h[d] = s;
}

// ---------------- G[:, 768:1024] = context * h_tilde ---------------------------------------------
__global__ void af_g4(const float* __restrict__ ctx, const float* __restrict__ h,
                      float* __restrict__ G)
{
    const int i = blockIdx.x * blockDim.x + threadIdx.x;   // over T*64 float4s
    const int t = i >> 6, c = (i & 63) * 4;
    const float4 cv = *(const float4*)(ctx + (size_t)t * DD + c);
    const float4 hv = *(const float4*)(h + c);
    const float4 o  = make_float4(cv.x * hv.x, cv.y * hv.y, cv.z * hv.z, cv.w * hv.w);
    *(float4*)(G + (size_t)t * (4 * DD) + 3 * DD + c) = o;
}

extern "C" void kernel_launch(void* const* d_in, const int* in_sizes, int n_in,
                              void* d_out, int out_size, void* d_ws, size_t ws_size,
                              hipStream_t stream)
{
    const float* ctx   = (const float*)d_in[0];
    const float* query = (const float*)d_in[1];
    const float* w     = (const float*)d_in[2];
    float* G = (float*)d_out;

    char* ws = (char*)d_ws;
    _Float16* qf16  = (_Float16*)(ws);                        // 1024*288*2 = 576 KB
    _Float16* qTf16 = (_Float16*)(ws + 589824);               // 512 KB
    float* b_buf    = (float*)(ws + 589824 + 524288);         // 32 KB
    float* partials = (float*)(ws + 589824 + 524288 + 32768); // 64 KB
    float* h        = (float*)(ws + 589824 + 524288 + 32768 + 65536); // 1 KB

    af_prep   <<<JD, DD, 0, stream>>>(query, w, qf16, qTf16);
    af_phase1 <<<TD / 16, 32, 0, stream>>>(ctx, w, qf16, qTf16, G, b_buf);
    af_hpart  <<<TD / 128, DD, 0, stream>>>(ctx, b_buf, partials);
    af_hreduce<<<1, DD, 0, stream>>>(partials, h);
    af_g4     <<<TD * DD / 4 / 256, 256, 0, stream>>>(ctx, h, G);
}
